// RegionalVFModel_15917148799015
// MI455X (gfx1250) — matrix operand order repaired and checked by reading, not compiled, not run on hardware
//
#include <hip/hip_runtime.h>

// ---------------- WMMA types (gfx1250, wave32) ----------------
typedef __attribute__((ext_vector_type(16))) __bf16 v16bf;
typedef __attribute__((ext_vector_type(8)))  __bf16 v8bf;
typedef __attribute__((ext_vector_type(8)))  float  v8f;

// ---------------- constants: sector scatter columns ----------------
// sector order: 0=sup_nasal(14), 1=sup_temporal(12), 2=inf_nasal(14), 3=inf_temporal(12)
__device__ __constant__ int HCOLS[4][14] = {
    {0,1,4,5,6,10,11,12,13,18,19,20,21,22},
    {2,3,7,8,9,14,15,16,17,23,24,25,-1,-1},
    {26,27,28,29,30,34,35,36,37,42,43,44,48,49},
    {31,32,33,38,39,40,41,45,46,47,50,51,-1,-1}};
__device__ __constant__ int HNPT[4] = {14,12,14,12};
// quadrant q (0:supN 1:supT 2:infN 3:infT retina) -> sector (MAP_OD)
__device__ __constant__ int QSEC[4] = {3,2,1,0};

__device__ __forceinline__ float gelu_exact(float v) {
    return 0.5f * v * (1.0f + erff(v * 0.70710678118654752440f));
}

// ---------------- kernel: quadrant mean + cls concat -> bf16 hi ----------------
// latent: (1024,197,1024) f32.  hi: (4 sectors, 1024, 2048) bf16
__global__ void rvf_quad_mean(const float* __restrict__ latent, __bf16* __restrict__ hi) {
    int b = blockIdx.x;
    int q = blockIdx.y;
    int s = QSEC[q];
    int R0 = (q >= 2) ? 7 : 0;
    int C0 = (q == 0 || q == 2) ? 7 : 0;
    const float* base = latent + (size_t)b * 197 * 1024;
    __bf16* out = hi + ((size_t)s * 1024 + b) * 2048;
    for (int d = threadIdx.x; d < 1024; d += blockDim.x) {
        float sum = 0.0f;
        #pragma unroll
        for (int pr = 0; pr < 7; ++pr) {
            int rowp = (R0 + pr) * 14 + C0;
            #pragma unroll
            for (int pc = 0; pc < 7; ++pc) {
                sum += base[(size_t)(1 + rowp + pc) * 1024 + d];
            }
        }
        out[d]        = (__bf16)(sum * (1.0f / 49.0f));
        out[1024 + d] = (__bf16)base[d];   // cls token
    }
}

// ---------------- kernel: W (fi x fo) f32 -> Wt (Npad x Kpad) bf16 (transposed, zero padded) ----
__global__ void rvf_convT_w(const float* __restrict__ W, __bf16* __restrict__ Wt,
                            int fi, int fo, int Kpad, int Npad) {
    int idx = blockIdx.x * blockDim.x + threadIdx.x;
    int total = Npad * Kpad;
    if (idx >= total) return;
    int n = idx / Kpad, k = idx % Kpad;
    float v = (n < fo && k < fi) ? W[(size_t)k * fo + n] : 0.0f;
    Wt[idx] = (__bf16)v;
}

// ---------------- kernel: X (rows x W, stride ldin) f32 -> Y (rows x Kpad) bf16 zero-padded ----
__global__ void rvf_f32_to_bf16(const float* __restrict__ X, int rows, int Wd, int ldin,
                                __bf16* __restrict__ Y, int Kpad) {
    int idx = blockIdx.x * blockDim.x + threadIdx.x;
    int total = rows * Kpad;
    if (idx >= total) return;
    int r = idx / Kpad, k = idx % Kpad;
    Y[idx] = (__bf16)((k < Wd) ? X[(size_t)r * ldin + k] : 0.0f);
}

// ---------------- kernel: generic WMMA GEMM  C = A(MxK) * Bt(NpadxK)^T + bias ----------------
// one wave (32 lanes) per 16x16 tile; K multiple of 32; bf16 inputs, f32 accumulate.
__global__ void rvf_wmma_gemm(const __bf16* __restrict__ A, const __bf16* __restrict__ Bt,
                              const float* __restrict__ bias, int nbias,
                              float* __restrict__ C, int K, int Nld) {
    int tileM = blockIdx.x, tileN = blockIdx.y;
    int lane  = threadIdx.x & 31;
    int half  = lane >> 4;            // 0: lanes 0-15, 1: lanes 16-31
    int r16   = lane & 15;
    // A frag: lane half 0 holds K {0..7,16..23}, half 1 holds K {8..15,24..31}
    const __bf16* aRow = A  + (size_t)(tileM * 16 + r16) * K + half * 8;
    // B frag: lane = column; half 0 holds K {0..7,16..23}, half 1 holds K {8..15,24..31}
    const __bf16* bRow = Bt + (size_t)(tileN * 16 + r16) * K + half * 8;
    v8f acc = {0.f,0.f,0.f,0.f,0.f,0.f,0.f,0.f};
    for (int k0 = 0; k0 < K; k0 += 32) {
        union { v16bf v; v8bf h[2]; } af, bfv;
        af.h[0]  = *(const v8bf*)(aRow + k0);
        af.h[1]  = *(const v8bf*)(aRow + k0 + 16);
        bfv.h[0] = *(const v8bf*)(bRow + k0);
        bfv.h[1] = *(const v8bf*)(bRow + k0 + 16);
        acc = __builtin_amdgcn_wmma_f32_16x16x32_bf16(
            /*neg_a=*/false, af.v, /*neg_b=*/false, bfv.v,
            /*c_mod=*/(short)0, acc, /*reuse_a=*/false, /*reuse_b=*/false);
    }
    int col = tileN * 16 + r16;
    float bz = (col < nbias) ? bias[col] : 0.0f;
    int rowBase = tileM * 16 + half * 8;   // VGPR i: M = rowBase + i
    #pragma unroll
    for (int i = 0; i < 8; ++i) {
        C[(size_t)(rowBase + i) * Nld + col] = acc[i] + bz;
    }
}

// ---------------- kernel: per-row LayerNorm + GELU -> bf16 (zero padded to Kpad) -------------
__global__ void rvf_ln_gelu(const float* __restrict__ X, int Wd, int ldin,
                            const float* __restrict__ g, const float* __restrict__ be,
                            __bf16* __restrict__ Y, int Kpad) {
    int row = blockIdx.x;
    int t   = threadIdx.x;           // 256 threads
    __shared__ float s1[256], s2[256];
    float sum = 0.f, sq = 0.f;
    for (int k = t; k < Wd; k += 256) {
        float v = X[(size_t)row * ldin + k];
        sum += v; sq += v * v;
    }
    s1[t] = sum; s2[t] = sq;
    __syncthreads();
    for (int o = 128; o > 0; o >>= 1) {
        if (t < o) { s1[t] += s1[t + o]; s2[t] += s2[t + o]; }
        __syncthreads();
    }
    float mean = s1[0] / (float)Wd;
    float var  = s2[0] / (float)Wd - mean * mean;
    float rstd = rsqrtf(var + 1e-5f);
    for (int k = t; k < Kpad; k += 256) {
        float o = 0.0f;
        if (k < Wd) {
            float v = (X[(size_t)row * ldin + k] - mean) * rstd * g[k] + be[k];
            o = gelu_exact(v);
        }
        Y[(size_t)row * Kpad + k] = (__bf16)o;
    }
}

// ---------------- kernel: head epilogue: LN(64)+GELU+ (64 x npt) proj, scatter into pred ------
__global__ void rvf_head_epilogue(const float* __restrict__ hpre,   // (1024,64) this sector
                                  const float* __restrict__ g, const float* __restrict__ be,
                                  const float* __restrict__ W2, const float* __restrict__ b2,
                                  int s, float* __restrict__ pred) {
    int b = blockIdx.x;
    int t = threadIdx.x;   // 64 threads
    __shared__ float red[64];
    __shared__ float sh[64];
    float x = hpre[(size_t)b * 64 + t];
    red[t] = x;
    __syncthreads();
    for (int o = 32; o > 0; o >>= 1) { if (t < o) red[t] += red[t + o]; __syncthreads(); }
    float mean = red[0] * (1.0f / 64.0f);
    __syncthreads();
    float d = x - mean;
    red[t] = d * d;
    __syncthreads();
    for (int o = 32; o > 0; o >>= 1) { if (t < o) red[t] += red[t + o]; __syncthreads(); }
    float rstd = rsqrtf(red[0] * (1.0f / 64.0f) + 1e-5f);
    float h = d * rstd * g[t] + be[t];
    sh[t] = gelu_exact(h);
    __syncthreads();
    int npt = HNPT[s];
    if (t < npt) {
        float acc = b2[t];
        #pragma unroll 8
        for (int i = 0; i < 64; ++i) acc += sh[i] * W2[i * npt + t];
        pred[(size_t)b * 52 + HCOLS[s][t]] = acc;
    }
}

// ---------------- kernel: pred += alpha * ffout ----------------
__global__ void rvf_fusion_add(float* __restrict__ pred, const float* __restrict__ ffout,
                               const float* __restrict__ alpha) {
    int idx = blockIdx.x * blockDim.x + threadIdx.x;
    if (idx >= 1024 * 52) return;
    int b = idx / 52, j = idx % 52;
    pred[idx] += alpha[0] * ffout[(size_t)b * 64 + j];
}

// ---------------- kernel: out = clip(threshold(h5 + rw*pred)) ----------------
__global__ void rvf_final(const float* __restrict__ h5, const float* __restrict__ pred,
                          const float* __restrict__ rw, float* __restrict__ out) {
    int idx = blockIdx.x * blockDim.x + threadIdx.x;
    if (idx >= 1024 * 52) return;
    int b = idx / 52, j = idx % 52;
    float v = h5[(size_t)b * 64 + j] + rw[0] * pred[idx];
    v = (v < 0.1f) ? 0.0f : v;
    v = fminf(v, 35.0f);
    out[idx] = v;
}

// ===================== host orchestration =====================
extern "C" void kernel_launch(void* const* d_in, const int* in_sizes, int n_in,
                              void* d_out, int out_size, void* d_ws, size_t ws_size,
                              hipStream_t stream) {
    (void)in_sizes; (void)n_in; (void)out_size; (void)ws_size;
    const float* latent = (const float*)d_in[0];
    // heads: sector s at base 1 + 6*s : W1,b1,g,be,W2,b2
    // fusion: 25..31 ; decoder: 32..50 ; average_multi: 51
    char* ws = (char*)d_ws;
    // -------- workspace layout (all offsets multiples of 2048 bytes) --------
    const size_t OFF_HI    = 0;                               // 4*1024*2048 bf16 = 16 MB
    const size_t OFF_W1T   = OFF_HI    + 4ull*1024*2048*2;    // 4*64*2048 bf16
    const size_t OFF_HPRE  = OFF_W1T   + 4ull*64*2048*2;      // 4*1024*64 f32
    const size_t OFF_PRED  = OFF_HPRE  + 4ull*1024*64*4;      // 1024*52 f32
    const size_t OFF_PREDB = OFF_PRED  + 1024ull*52*4;        // 1024*64 bf16
    const size_t OFF_FW1T  = OFF_PREDB + 1024ull*64*2;        // 112*64 bf16
    const size_t OFF_FPRE  = OFF_FW1T  + 112ull*64*2;         // 1024*112 f32
    const size_t OFF_FHB   = OFF_FPRE  + 1024ull*112*4;       // 1024*128 bf16
    const size_t OFF_FW2T  = OFF_FHB   + 1024ull*128*2;       // 64*128 bf16
    const size_t OFF_FFOUT = OFF_FW2T  + 64ull*128*2;         // 1024*64 f32
    const size_t OFF_WT1   = OFF_FFOUT + 1024ull*64*4;        // 256*64 bf16
    const size_t OFF_WT2   = OFF_WT1   + 256ull*64*2;         // 512*256 bf16
    const size_t OFF_WT3   = OFF_WT2   + 512ull*256*2;        // 512*512 bf16
    const size_t OFF_WT4   = OFF_WT3   + 512ull*512*2;        // 256*512 bf16
    const size_t OFF_WT5   = OFF_WT4   + 256ull*512*2;        // 64*256 bf16
    const size_t OFF_PREA  = OFF_WT5   + 64ull*256*2;         // 1024*512 f32
    const size_t OFF_PREB  = OFF_PREA  + 1024ull*512*4;       // 1024*512 f32
    const size_t OFF_ACTA  = OFF_PREB  + 1024ull*512*4;       // 1024*512 bf16
    const size_t OFF_ACTB  = OFF_ACTA  + 1024ull*512*2;       // 1024*512 bf16

    __bf16* hi    = (__bf16*)(ws + OFF_HI);
    __bf16* w1t   = (__bf16*)(ws + OFF_W1T);
    float*  hpre  = (float*) (ws + OFF_HPRE);
    float*  pred  = (float*) (ws + OFF_PRED);
    __bf16* predb = (__bf16*)(ws + OFF_PREDB);
    __bf16* fw1t  = (__bf16*)(ws + OFF_FW1T);
    float*  fpre  = (float*) (ws + OFF_FPRE);
    __bf16* fhb   = (__bf16*)(ws + OFF_FHB);
    __bf16* fw2t  = (__bf16*)(ws + OFF_FW2T);
    float*  ffout = (float*) (ws + OFF_FFOUT);
    __bf16* wt1   = (__bf16*)(ws + OFF_WT1);
    __bf16* wt2   = (__bf16*)(ws + OFF_WT2);
    __bf16* wt3   = (__bf16*)(ws + OFF_WT3);
    __bf16* wt4   = (__bf16*)(ws + OFF_WT4);
    __bf16* wt5   = (__bf16*)(ws + OFF_WT5);
    float*  preA  = (float*) (ws + OFF_PREA);
    float*  preB  = (float*) (ws + OFF_PREB);
    __bf16* actAb = (__bf16*)(ws + OFF_ACTA);
    __bf16* actBb = (__bf16*)(ws + OFF_ACTB);

    auto cdiv = [](long long a, long long b){ return (int)((a + b - 1) / b); };

    // 1) streaming pass over latent (HBM-bound: 826 MB, read-once, coalesced)
    rvf_quad_mean<<<dim3(1024, 4), 256, 0, stream>>>(latent, hi);

    // 2) head weight transpose + WMMA GEMM (1024x2048 @ 2048x64) + epilogue, per sector
    for (int s = 0; s < 4; ++s) {
        const float* W1 = (const float*)d_in[1 + 6*s + 0];
        const float* b1 = (const float*)d_in[1 + 6*s + 1];
        const float* g  = (const float*)d_in[1 + 6*s + 2];
        const float* be = (const float*)d_in[1 + 6*s + 3];
        const float* W2 = (const float*)d_in[1 + 6*s + 4];
        const float* b2 = (const float*)d_in[1 + 6*s + 5];
        __bf16* wts = w1t + (size_t)s * 64 * 2048;
        float*  hps = hpre + (size_t)s * 1024 * 64;
        rvf_convT_w<<<cdiv(64*2048, 256), 256, 0, stream>>>(W1, wts, 2048, 64, 2048, 64);
        rvf_wmma_gemm<<<dim3(64, 4), 32, 0, stream>>>(hi + (size_t)s*1024*2048, wts, b1, 64,
                                                      hps, 2048, 64);
        rvf_head_epilogue<<<1024, 64, 0, stream>>>(hps, g, be, W2, b2, s, pred);
    }

    // 3) fusion: pred(52) -> 104 -> 52, residual with alpha
    const float* fW1 = (const float*)d_in[25];
    const float* fb1 = (const float*)d_in[26];
    const float* fg  = (const float*)d_in[27];
    const float* fbe = (const float*)d_in[28];
    const float* fW2 = (const float*)d_in[29];
    const float* fb2 = (const float*)d_in[30];
    const float* alpha = (const float*)d_in[31];
    rvf_f32_to_bf16<<<cdiv(1024*64, 256), 256, 0, stream>>>(pred, 1024, 52, 52, predb, 64);
    rvf_convT_w<<<cdiv(112*64, 256), 256, 0, stream>>>(fW1, fw1t, 52, 104, 64, 112);
    rvf_wmma_gemm<<<dim3(64, 7), 32, 0, stream>>>(predb, fw1t, fb1, 104, fpre, 64, 112);
    rvf_ln_gelu<<<1024, 256, 0, stream>>>(fpre, 104, 112, fg, fbe, fhb, 128);
    rvf_convT_w<<<cdiv(64*128, 256), 256, 0, stream>>>(fW2, fw2t, 104, 52, 128, 64);
    rvf_wmma_gemm<<<dim3(64, 4), 32, 0, stream>>>(fhb, fw2t, fb2, 52, ffout, 128, 64);
    rvf_fusion_add<<<cdiv(1024*52, 256), 256, 0, stream>>>(pred, ffout, alpha);

    // 4) decoder 52 -> 256 -> 512 -> 512 -> 256 -> 52 (all WMMA)
    const float* dW1 = (const float*)d_in[32]; const float* db1 = (const float*)d_in[33];
    const float* dg1 = (const float*)d_in[34]; const float* dbe1= (const float*)d_in[35];
    const float* dW2 = (const float*)d_in[36]; const float* db2 = (const float*)d_in[37];
    const float* dg2 = (const float*)d_in[38]; const float* dbe2= (const float*)d_in[39];
    const float* dW3 = (const float*)d_in[40]; const float* db3 = (const float*)d_in[41];
    const float* dg3 = (const float*)d_in[42]; const float* dbe3= (const float*)d_in[43];
    const float* dW4 = (const float*)d_in[44]; const float* db4 = (const float*)d_in[45];
    const float* dg4 = (const float*)d_in[46]; const float* dbe4= (const float*)d_in[47];
    const float* dW5 = (const float*)d_in[48]; const float* db5 = (const float*)d_in[49];
    const float* rw  = (const float*)d_in[50];

    rvf_f32_to_bf16<<<cdiv(1024*64, 256), 256, 0, stream>>>(pred, 1024, 52, 52, predb, 64);

    rvf_convT_w<<<cdiv(256*64, 256), 256, 0, stream>>>(dW1, wt1, 52, 256, 64, 256);
    rvf_wmma_gemm<<<dim3(64, 16), 32, 0, stream>>>(predb, wt1, db1, 256, preA, 64, 256);
    rvf_ln_gelu<<<1024, 256, 0, stream>>>(preA, 256, 256, dg1, dbe1, actAb, 256);

    rvf_convT_w<<<cdiv(512*256, 256), 256, 0, stream>>>(dW2, wt2, 256, 512, 256, 512);
    rvf_wmma_gemm<<<dim3(64, 32), 32, 0, stream>>>(actAb, wt2, db2, 512, preB, 256, 512);
    rvf_ln_gelu<<<1024, 256, 0, stream>>>(preB, 512, 512, dg2, dbe2, actBb, 512);

    rvf_convT_w<<<cdiv(512*512, 256), 256, 0, stream>>>(dW3, wt3, 512, 512, 512, 512);
    rvf_wmma_gemm<<<dim3(64, 32), 32, 0, stream>>>(actBb, wt3, db3, 512, preA, 512, 512);
    rvf_ln_gelu<<<1024, 256, 0, stream>>>(preA, 512, 512, dg3, dbe3, actAb, 512);

    rvf_convT_w<<<cdiv(256*512, 256), 256, 0, stream>>>(dW4, wt4, 512, 256, 512, 256);
    rvf_wmma_gemm<<<dim3(64, 16), 32, 0, stream>>>(actAb, wt4, db4, 256, preB, 512, 256);
    rvf_ln_gelu<<<1024, 256, 0, stream>>>(preB, 256, 256, dg4, dbe4, actBb, 256);

    rvf_convT_w<<<cdiv(64*256, 256), 256, 0, stream>>>(dW5, wt5, 256, 52, 256, 64);
    rvf_wmma_gemm<<<dim3(64, 4), 32, 0, stream>>>(actBb, wt5, db5, 52, preA, 256, 64);

    // 5) residual + threshold + clip -> d_out (1024 x 52 f32)
    rvf_final<<<cdiv(1024*52, 256), 256, 0, stream>>>(preA, pred, rw, (float*)d_out);
}
